// GNNModel_25924422599371
// MI455X (gfx1250) — matrix operand order repaired and checked by reading, not compiled, not run on hardware
//
#include <hip/hip_runtime.h>
#include <hip/hip_bf16.h>
#include <stdint.h>

// ---------------------------------------------------------------------------
// Problem constants (match reference)
// ---------------------------------------------------------------------------
#define G_GRAPHS 128
#define N0_NODES 256
#define NT_MAX   (G_GRAPHS * N0_NODES)   // 32768
#define E_EDGES  (NT_MAX * 8)            // 262144
#define D_HID    512
#define H_HEADS  4
#define D_OUT    128
#define NEG_BIG  (-3.0e38f)

typedef __attribute__((ext_vector_type(16))) __bf16 v16bf;
typedef __attribute__((ext_vector_type(8)))  float  v8f;
typedef float        f32x4 __attribute__((ext_vector_type(4)));
typedef float        f32x2 __attribute__((ext_vector_type(2)));
typedef unsigned int u32x4 __attribute__((ext_vector_type(4)));
typedef int          i32x4 __attribute__((ext_vector_type(4)));
typedef int          i32x8 __attribute__((ext_vector_type(8)));

#if __has_builtin(__builtin_amdgcn_tensor_load_to_lds) && \
    __has_builtin(__builtin_amdgcn_s_wait_tensorcnt)
#define USE_TDM 1
#endif

__device__ __forceinline__ __bf16 f2bf(float f) {
    unsigned u = __float_as_uint(f);
    unsigned r = u + 0x7fffu + ((u >> 16) & 1u);     // round-to-nearest-even
    unsigned short h = (unsigned short)(r >> 16);
    return __builtin_bit_cast(__bf16, h);
}

__device__ __forceinline__ float lrelu02(float x) { return x > 0.0f ? x : 0.2f * x; }

// order-preserving float <-> uint key (for atomicMax on floats incl. negatives)
__device__ __forceinline__ unsigned fenc(float f) {
    unsigned u = __float_as_uint(f);
    return (u >> 31) ? ~u : (u | 0x80000000u);
}
__device__ __forceinline__ float fdec(unsigned k) {
    return (k & 0x80000000u) ? __uint_as_float(k ^ 0x80000000u)
                             : __uint_as_float(~k);
}

#ifdef USE_TDM
// Issue a TDM DMA of a 16x32 fp32 tile (row stride = lda elements) into LDS.
// D# fields per CDNA5 ISA ch.8: group0 = {count, lds_addr, global_addr, type};
// group1 = {data_size=4B, tensor dims (huge: tile always in-bounds),
//           tile_dim0=32, tile_dim1=16, dim0 stride = lda}.
// This toolchain's builtin takes 6 args: (g0, g1, g2, g3, g_extra, cpol).
__device__ __forceinline__ void tdm_issue_a_tile(unsigned lds_off,
                                                 const float* gptr, int lda)
{
    unsigned long long ga = (unsigned long long)(uintptr_t)gptr;
    u32x4 g0;
    g0[0] = 1u;                                            // count = 1 (valid)
    g0[1] = lds_off;                                       // lds_addr
    g0[2] = (unsigned)ga;                                  // global_addr[31:0]
    g0[3] = (unsigned)((ga >> 32) & 0x01ffffffu)           // global_addr[56:32]
            | (2u << 30);                                  // type = 2 ("image")
    i32x8 g1;
    g1[0] = (int)(2u << 16);                               // data_size = 4B
    g1[1] = (int)(0xffffu << 16);                          // tensor_dim0 lo16
    g1[2] = (int)(0x7fffu | (0xffffu << 16));              // dim0 hi | dim1 lo
    g1[3] = (int)(0x7fffu | (32u << 16));                  // dim1 hi | tile_dim0=32
    g1[4] = 16;                                            // tile_dim1=16, tile_dim2=0
    g1[5] = lda;                                           // tensor_dim0_stride lo32
    g1[6] = 0;
    g1[7] = 0;
    i32x4 gz4 = {0, 0, 0, 0};
    i32x8 gz8 = {0, 0, 0, 0, 0, 0, 0, 0};
    __builtin_amdgcn_tensor_load_to_lds(g0, g1, gz4, gz4, gz8, 0);
}
#endif

// ---------------------------------------------------------------------------
// WMMA bf16 GEMM: C[M,N] = act(A[M,K] * B[K,N] + bias)
// block = 256 threads = 8 waves; block tile = 16 rows x 128 cols
// grid = (M/16, N/128); K % 32 == 0; lda/ldb multiples of 4, col0 mult of 16
// A tile staged fp32 in LDS via TDM (double-buffered), converted to bf16 in
// the fragment gather; B fragments gathered as 16 strided fp32 loads per lane.
// ---------------------------------------------------------------------------
__global__ void k_wmma_gemm(const float* __restrict__ A, int lda,
                            const float* __restrict__ B, int ldb,
                            const float* __restrict__ bias,
                            float* __restrict__ C, int ldc,
                            int K, int relu)
{
    __shared__ float Asf[2][16 * 32];
    const int tid  = threadIdx.x;
    const int wave = tid >> 5;
    const int lane = tid & 31;
    const int row0 = blockIdx.x * 16;
    const int col0 = blockIdx.y * 128 + wave * 16;
    const int half = lane >> 4;
    const int mrow = lane & 15;        // A fragment row / C column index
    const int nk   = K >> 5;

    v8f acc = {};

#ifdef USE_TDM
    const float* Abase = A + (size_t)row0 * lda;
    if (wave == 0)
        tdm_issue_a_tile((unsigned)(uintptr_t)&Asf[0][0], Abase, lda);
#endif

    for (int t = 0; t < nk; ++t) {
        const int k0 = t << 5;
        const float* buf = Asf[t & 1];

#ifdef USE_TDM
        if (wave == 0) {
            if (t + 1 < nk) {
                tdm_issue_a_tile((unsigned)(uintptr_t)&Asf[(t + 1) & 1][0],
                                 Abase + (k0 + 32), lda);
                __builtin_amdgcn_s_wait_tensorcnt(1);   // tile t landed
            } else {
                __builtin_amdgcn_s_wait_tensorcnt(0);
            }
        }
#else
        {   // cooperative fallback: each thread copies 2 consecutive floats
            int e = tid * 2;
            int r = e >> 5, kk = e & 31;
            f32x2 v = *(const f32x2*)(A + (size_t)(row0 + r) * lda + k0 + kk);
            *(f32x2*)&Asf[t & 1][e] = v;
        }
#endif
        __syncthreads();

        // A fragment: lane(half,mrow); K runs {8*half..+7} and {16+8*half..+7}
        const float* ab = buf + mrow * 32;
        const f32x4* Ap = (const f32x4*)(ab + 8 * half);
        const f32x4* Aq = (const f32x4*)(ab + 16 + 8 * half);
        f32x4 a0 = Ap[0], a1 = Ap[1], a2 = Aq[0], a3 = Aq[1];
        v16bf fa;
        fa[0]  = f2bf(a0[0]); fa[1]  = f2bf(a0[1]); fa[2]  = f2bf(a0[2]); fa[3]  = f2bf(a0[3]);
        fa[4]  = f2bf(a1[0]); fa[5]  = f2bf(a1[1]); fa[6]  = f2bf(a1[2]); fa[7]  = f2bf(a1[3]);
        fa[8]  = f2bf(a2[0]); fa[9]  = f2bf(a2[1]); fa[10] = f2bf(a2[2]); fa[11] = f2bf(a2[3]);
        fa[12] = f2bf(a3[0]); fa[13] = f2bf(a3[1]); fa[14] = f2bf(a3[2]); fa[15] = f2bf(a3[3]);

        // B fragment: lane(half,mrow) = N column; K runs {8*half..+7} and {16+8*half..+7}
        const float* Brow = B + (size_t)(k0 + 8 * half) * ldb + col0 + mrow;
        v16bf fb;
        #pragma unroll
        for (int q = 0; q < 8; ++q) {
            fb[q]     = f2bf(Brow[(size_t)q * ldb]);
            fb[8 + q] = f2bf(Brow[(size_t)(16 + q) * ldb]);
        }
        // speculative prefetch of next K-step's B row (safe past end)
        __builtin_prefetch(Brow + (size_t)32 * ldb, 0, 1);

        acc = __builtin_amdgcn_wmma_f32_16x16x32_bf16(
                  false, fa, false, fb, (short)0, acc, false, false);
        __syncthreads();
    }

    // C layout: VGPR r -> M = r + 8*half ; N = lane & 15
    float bv = bias ? bias[col0 + mrow] : 0.0f;
    #pragma unroll
    for (int r = 0; r < 8; ++r) {
        float v = acc[r] + bv;
        if (relu) v = fmaxf(v, 0.0f);
        C[(size_t)(row0 + half * 8 + r) * ldc + col0 + mrow] = v;
    }
}

// ---------------------------------------------------------------------------
// Layer-0 GEMM, K=3 (memory bound, no WMMA): H[n,c] = sum_k x[n,k]*W[k,c]
// ---------------------------------------------------------------------------
__global__ void k_gemm_k3(const float* __restrict__ x, const float* __restrict__ W,
                          float* __restrict__ Hb, int NT)
{
    int idx = blockIdx.x * blockDim.x + threadIdx.x;
    if (idx >= NT * D_HID) return;
    int n = idx >> 9, c = idx & 511;
    Hb[idx] = x[n * 3 + 0] * W[c] + x[n * 3 + 1] * W[512 + c] + x[n * 3 + 2] * W[1024 + c];
}

// ---------------------------------------------------------------------------
// attention pre-scores al/ar per (node, head)
// ---------------------------------------------------------------------------
__global__ void k_alar(const float* __restrict__ Hb,
                       const float* __restrict__ asrc, const float* __restrict__ adst,
                       float* __restrict__ AL, float* __restrict__ AR, int NT)
{
    int idx = blockIdx.x * blockDim.x + threadIdx.x;
    if (idx >= NT * H_HEADS) return;
    int n = idx >> 2, h = idx & 3;
    const float* hp = Hb + (size_t)n * D_HID + h * D_OUT;
    const float* as = asrc + h * D_OUT;
    const float* ad = adst + h * D_OUT;
    float al = 0.0f, ar = 0.0f;
    for (int c = 0; c < D_OUT; ++c) { float f = hp[c]; al += f * as[c]; ar += f * ad[c]; }
    AL[idx] = al; AR[idx] = ar;
}

__global__ void k_init_m(const float* __restrict__ AL, const float* __restrict__ AR,
                         unsigned* __restrict__ MH, int NT)
{
    int idx = blockIdx.x * blockDim.x + threadIdx.x;
    if (idx >= NT * H_HEADS) return;
    MH[idx] = fenc(lrelu02(AL[idx] + AR[idx]));   // init with self-loop score
}

__global__ void k_edge_max(const int* __restrict__ SRC, const int* __restrict__ DST,
                           const int* __restrict__ MASK,
                           const float* __restrict__ AL, const float* __restrict__ AR,
                           unsigned* __restrict__ MH)
{
    int idx = blockIdx.x * blockDim.x + threadIdx.x;
    if (idx >= E_EDGES * H_HEADS) return;
    int e = idx >> 2, h = idx & 3;
    if (!MASK[e]) return;
    int s = SRC[e], d = DST[e];
    float v = lrelu02(AL[s * 4 + h] + AR[d * 4 + h]);
    atomicMax(&MH[d * 4 + h], fenc(v));
}

__global__ void k_init_numden(const float* __restrict__ Hb,
                              const float* __restrict__ AL, const float* __restrict__ AR,
                              const unsigned* __restrict__ MH,
                              float* __restrict__ NUM, float* __restrict__ DEN, int NT)
{
    int idx = blockIdx.x * blockDim.x + threadIdx.x;
    if (idx >= NT * D_HID) return;
    int n = idx >> 9, c = idx & 511, h = c >> 7;
    float m  = fdec(MH[n * 4 + h]);
    float ws = __expf(lrelu02(AL[n * 4 + h] + AR[n * 4 + h]) - m);
    NUM[idx] = ws * Hb[idx];
    if ((c & 127) == 0) DEN[n * 4 + h] = ws;
}

// one block per edge; 256 threads cover 512 channels
__global__ void k_edge_scatter(const int* __restrict__ SRC, const int* __restrict__ DST,
                               const int* __restrict__ MASK,
                               const float* __restrict__ AL, const float* __restrict__ AR,
                               const unsigned* __restrict__ MH,
                               const float* __restrict__ Hb,
                               float* __restrict__ NUM, float* __restrict__ DEN)
{
    int e = blockIdx.x;
    if (!MASK[e]) return;
    int s = SRC[e], d = DST[e];
    for (int c = threadIdx.x; c < D_HID; c += 256) {
        int h = c >> 7;
        float m = fdec(MH[d * 4 + h]);
        float w = __expf(lrelu02(AL[s * 4 + h] + AR[d * 4 + h]) - m);
        atomicAdd(&NUM[(size_t)d * D_HID + c], w * Hb[(size_t)s * D_HID + c]);
        if ((c & 127) == 0) atomicAdd(&DEN[d * 4 + h], w);
    }
}

__global__ void k_gat_final(const float* __restrict__ NUM, const float* __restrict__ DEN,
                            const float* __restrict__ bvec, float* __restrict__ Hb, int NT)
{
    int idx = blockIdx.x * blockDim.x + threadIdx.x;
    if (idx >= NT * D_HID) return;
    int n = idx >> 9, c = idx & 511, h = c >> 7;
    Hb[idx] = NUM[idx] / DEN[n * 4 + h] + bvec[c];
}

// ---------------------------------------------------------------------------
// BatchNorm (training-style biased var) + ReLU
// ---------------------------------------------------------------------------
__global__ void k_bn_stats(const float* __restrict__ Hb, int NT,
                           float* __restrict__ MU, float* __restrict__ RS)
{
    __shared__ float ssum[256], ssq[256];
    int c = blockIdx.x, tid = threadIdx.x;
    float s = 0.0f, q = 0.0f;
    for (int n = tid; n < NT; n += 256) {
        float v = Hb[(size_t)n * D_HID + c];
        s += v; q += v * v;
    }
    ssum[tid] = s; ssq[tid] = q;
    __syncthreads();
    for (int st = 128; st > 0; st >>= 1) {
        if (tid < st) { ssum[tid] += ssum[tid + st]; ssq[tid] += ssq[tid + st]; }
        __syncthreads();
    }
    if (tid == 0) {
        float mu  = ssum[0] / (float)NT;
        float var = ssq[0] / (float)NT - mu * mu;
        MU[c] = mu;
        RS[c] = rsqrtf(var + 1e-5f);
    }
}

__global__ void k_bn_apply(const float* __restrict__ Hb,
                           const float* __restrict__ MU, const float* __restrict__ RS,
                           const float* __restrict__ g, const float* __restrict__ b,
                           float* __restrict__ Y, int NT)
{
    int idx = blockIdx.x * blockDim.x + threadIdx.x;
    if (idx >= NT * D_HID) return;
    int c = idx & 511;
    float v = (Hb[idx] - MU[c]) * RS[c] * g[c] + b[c];
    Y[idx] = fmaxf(v, 0.0f);
}

// ---------------------------------------------------------------------------
// TopK pooling: per-graph block, iterative argmax (desc order, low-index ties)
// ---------------------------------------------------------------------------
__global__ void k_topk(const float* __restrict__ Xin, int n, int k,
                       const float* __restrict__ p,
                       float* __restrict__ NEWX, int* __restrict__ NEWPOS)
{
    __shared__ float sval[256];
    __shared__ float rv[256];
    __shared__ int   ri[256];
    __shared__ float s_gate;
    __shared__ int   s_bi;
    int g = blockIdx.x, tid = threadIdx.x;

    // ||p||
    float ps = 0.0f;
    for (int c = tid; c < D_HID; c += 256) ps += p[c] * p[c];
    rv[tid] = ps;
    __syncthreads();
    for (int st = 128; st > 0; st >>= 1) {
        if (tid < st) rv[tid] += rv[tid + st];
        __syncthreads();
    }
    float invn = rsqrtf(rv[0] + 1e-30f);
    __syncthreads();

    // scores
    if (tid < n) {
        const float* xr = Xin + (size_t)(g * n + tid) * D_HID;
        float dot = 0.0f;
        for (int c = 0; c < D_HID; ++c) dot += xr[c] * p[c];
        sval[tid] = dot * invn;
        NEWPOS[g * n + tid] = -1;
    } else {
        sval[tid] = NEG_BIG;
    }
    __syncthreads();

    for (int t = 0; t < k; ++t) {
        rv[tid] = sval[tid]; ri[tid] = tid;
        __syncthreads();
        for (int st = 128; st > 0; st >>= 1) {
            if (tid < st) {
                if (rv[tid + st] > rv[tid] ||
                    (rv[tid + st] == rv[tid] && ri[tid + st] < ri[tid])) {
                    rv[tid] = rv[tid + st]; ri[tid] = ri[tid + st];
                }
            }
            __syncthreads();
        }
        if (tid == 0) {
            s_bi = ri[0];
            s_gate = tanhf(rv[0]);
            NEWPOS[g * n + ri[0]] = g * k + t;
            sval[ri[0]] = NEG_BIG;
        }
        __syncthreads();
        int bi = s_bi; float gate = s_gate;
        const float* src = Xin + (size_t)(g * n + bi) * D_HID;
        float* dst = NEWX + (size_t)(g * k + t) * D_HID;
        for (int c = tid; c < D_HID; c += 256) dst[c] = src[c] * gate;
        __syncthreads();
    }
}

__global__ void k_edge_init(const int* __restrict__ es, const int* __restrict__ ed,
                            int* __restrict__ SRC, int* __restrict__ DST, int* __restrict__ MASK)
{
    int i = blockIdx.x * blockDim.x + threadIdx.x;
    if (i >= E_EDGES) return;
    SRC[i] = es[i]; DST[i] = ed[i]; MASK[i] = 1;
}

__global__ void k_remap(int* __restrict__ SRC, int* __restrict__ DST, int* __restrict__ MASK,
                        const int* __restrict__ NEWPOS)
{
    int i = blockIdx.x * blockDim.x + threadIdx.x;
    if (i >= E_EDGES) return;
    int ns = NEWPOS[SRC[i]], nd = NEWPOS[DST[i]];
    MASK[i] = (MASK[i] && ns >= 0 && nd >= 0) ? 1 : 0;
    SRC[i] = ns >= 0 ? ns : 0;
    DST[i] = nd >= 0 ? nd : 0;
}

// per-graph max & mean pooling -> [G, 1024]
__global__ void k_poolfeat(const float* __restrict__ X, int k, float* __restrict__ POOLED)
{
    int g = blockIdx.x, c = threadIdx.x;   // blockDim = 512
    float mx = NEG_BIG, sm = 0.0f;
    for (int j = 0; j < k; ++j) {
        float v = X[(size_t)(g * k + j) * D_HID + c];
        mx = fmaxf(mx, v); sm += v;
    }
    POOLED[(size_t)g * 1024 + c]       = mx;
    POOLED[(size_t)g * 1024 + 512 + c] = sm / (float)k;
}

// out[g] = dot(Z[g,:], Wf2) + bf2
__global__ void k_final_dot(const float* __restrict__ Z, const float* __restrict__ Wf2,
                            const float* __restrict__ bf2, float* __restrict__ out)
{
    int g = threadIdx.x;    // 128 threads
    float acc = bf2[0];
    for (int c = 0; c < D_OUT; ++c) acc += Z[(size_t)g * D_OUT + c] * Wf2[c];
    out[g] = acc;
}

// ---------------------------------------------------------------------------
// host orchestration
// ---------------------------------------------------------------------------
extern "C" void kernel_launch(void* const* d_in, const int* in_sizes, int n_in,
                              void* d_out, int out_size, void* d_ws, size_t ws_size,
                              hipStream_t stream)
{
    (void)in_sizes; (void)n_in; (void)out_size; (void)ws_size;

    const float* x_in = (const float*)d_in[0];
    const int*   es   = (const int*)d_in[1];
    const int*   ed   = (const int*)d_in[2];
    auto P = [&](int layer, int j) -> const float* {
        return (const float*)d_in[3 + layer * 13 + j];
    };
    // param j: 0=W 1=a_src 2=a_dst 3=b 4=bn_g 5=bn_b 6=pool_p
    //          7=Wq 8=bq 9=Wk 10=bk 11=Wv 12=bv
    const float* Wf1 = (const float*)d_in[42];
    const float* bf1 = (const float*)d_in[43];
    const float* Wf2 = (const float*)d_in[44];
    const float* bf2 = (const float*)d_in[45];

    size_t off = 0;
    auto alloc = [&](size_t bytes) -> void* {
        void* pp = (char*)d_ws + off;
        off = (off + bytes + 255) & ~(size_t)255;
        return pp;
    };
    float*    XA     = (float*)alloc((size_t)NT_MAX * D_HID * 4);  // pooled x (next layer in)
    float*    HB     = (float*)alloc((size_t)NT_MAX * D_HID * 4);  // GAT output h
    float*    NUM    = (float*)alloc((size_t)NT_MAX * D_HID * 4);  // num accum / BN'd x
    float*    AL     = (float*)alloc((size_t)NT_MAX * 4 * 4);
    float*    AR     = (float*)alloc((size_t)NT_MAX * 4 * 4);
    unsigned* MH     = (unsigned*)alloc((size_t)NT_MAX * 4 * 4);
    float*    DEN    = (float*)alloc((size_t)NT_MAX * 4 * 4);
    int*      SRC    = (int*)alloc((size_t)E_EDGES * 4);
    int*      DST    = (int*)alloc((size_t)E_EDGES * 4);
    int*      MASK   = (int*)alloc((size_t)E_EDGES * 4);
    int*      NEWPOS = (int*)alloc((size_t)NT_MAX * 4);
    float*    MU     = (float*)alloc(512 * 4);
    float*    RS     = (float*)alloc(512 * 4);
    float*    POOLED = (float*)alloc((size_t)G_GRAPHS * 1024 * 4);
    float*    FEATS  = (float*)alloc((size_t)G_GRAPHS * 3072 * 4);
    float*    ZB     = (float*)alloc((size_t)G_GRAPHS * D_OUT * 4);

    auto cdiv = [](int a, int b) { return (a + b - 1) / b; };
    const int npg[3] = {256, 205, 164};
    const int kpg[3] = {205, 164, 132};   // ceil(0.8*n)

    k_edge_init<<<cdiv(E_EDGES, 256), 256, 0, stream>>>(es, ed, SRC, DST, MASK);

    for (int i = 0; i < 3; ++i) {
        const int NT = G_GRAPHS * npg[i];

        // h = x @ W_i
        if (i == 0) {
            k_gemm_k3<<<cdiv(NT * D_HID, 256), 256, 0, stream>>>(x_in, P(0, 0), HB, NT);
        } else {
            k_wmma_gemm<<<dim3(NT / 16, D_HID / 128), 256, 0, stream>>>(
                XA, D_HID, P(i, 0), D_HID, nullptr, HB, D_HID, D_HID, 0);
        }

        // attention scores + segment softmax + scatter
        k_alar<<<cdiv(NT * 4, 256), 256, 0, stream>>>(HB, P(i, 1), P(i, 2), AL, AR, NT);
        k_init_m<<<cdiv(NT * 4, 256), 256, 0, stream>>>(AL, AR, MH, NT);
        k_edge_max<<<cdiv(E_EDGES * 4, 256), 256, 0, stream>>>(SRC, DST, MASK, AL, AR, MH);
        k_init_numden<<<cdiv(NT * D_HID, 256), 256, 0, stream>>>(HB, AL, AR, MH, NUM, DEN, NT);
        k_edge_scatter<<<E_EDGES, 256, 0, stream>>>(SRC, DST, MASK, AL, AR, MH, HB, NUM, DEN);
        k_gat_final<<<cdiv(NT * D_HID, 256), 256, 0, stream>>>(NUM, DEN, P(i, 3), HB, NT);

        // BN + ReLU -> NUM (reused as normalized features)
        k_bn_stats<<<512, 256, 0, stream>>>(HB, NT, MU, RS);
        k_bn_apply<<<cdiv(NT * D_HID, 256), 256, 0, stream>>>(HB, MU, RS, P(i, 4), P(i, 5), NUM, NT);

        // TopK pool -> XA (new x) + edge remap
        k_topk<<<G_GRAPHS, 256, 0, stream>>>(NUM, npg[i], kpg[i], P(i, 6), XA, NEWPOS);
        k_remap<<<cdiv(E_EDGES, 256), 256, 0, stream>>>(SRC, DST, MASK, NEWPOS);

        // readout: [max|mean] -> POOLED -> feats_i = POOLED @ Wv + bv
        k_poolfeat<<<G_GRAPHS, 512, 0, stream>>>(XA, kpg[i], POOLED);
        k_wmma_gemm<<<dim3(G_GRAPHS / 16, 1024 / 128), 256, 0, stream>>>(
            POOLED, 1024, P(i, 11), 1024, P(i, 12), FEATS + i * 1024, 3072, 1024, 0);
    }

    // z = relu(FEATS @ Wf1 + bf1);  out = z @ Wf2 + bf2
    k_wmma_gemm<<<dim3(G_GRAPHS / 16, 1), 256, 0, stream>>>(
        FEATS, 3072, Wf1, D_OUT, bf1, ZB, D_OUT, 3072, 1);
    k_final_dot<<<1, 128, 0, stream>>>(ZB, Wf2, bf2, (float*)d_out);
}
